// MultiHeadAttention_19868518711593
// MI455X (gfx1250) — matrix operand order repaired and checked by reading, not compiled, not run on hardware
//
#include <hip/hip_runtime.h>
#include <hip/hip_bf16.h>

typedef __attribute__((ext_vector_type(16))) _Float16 v16h;
typedef __attribute__((ext_vector_type(8)))  _Float16 v8h;
typedef __attribute__((ext_vector_type(4)))  _Float16 v4h;
typedef __attribute__((ext_vector_type(8)))  float    v8f;
typedef __attribute__((ext_vector_type(4)))  float    v4f;

#define WMMA_F16(a, b, c) \
  __builtin_amdgcn_wmma_f32_16x16x32_f16(false, (a), false, (b), (short)0, (c), false, false)

// ---------------------------------------------------------------------------
// Fragment loaders (wave32, 16x16x32 f16 WMMA).
// A 16x32 (MxK): lane holds row m = lane&15; halves 0..7 = K kb..kb+7,
//                halves 8..15 = K 16+kb..16+kb+7, kb = (lane>>4)*8.
// B 32x16 (KxN): lane holds col n = lane&15; halves 0..7 = K kb..kb+7,
//                halves 8..15 = K 16+kb..16+kb+7, kb = (lane>>4)*8.  Source stored "n-major": src[n][k].
// C/D 16x16:     lane holds col n = lane&15; VGPR v holds row m = v + 8*(lane>>4).
// ---------------------------------------------------------------------------
static __device__ inline v16h load_a_frag(const _Float16* src, int stride, int k0, int lane) {
  int m  = lane & 15;
  int kb = (lane >> 4) * 8;
  union { v16h f; v8h h[2]; } u;
  u.h[0] = *(const v8h*)(src + m * stride + k0 + kb);
  u.h[1] = *(const v8h*)(src + m * stride + k0 + 16 + kb);
  return u.f;
}

static __device__ inline v16h load_b_frag(const _Float16* srcN, int stride, int k0, int lane) {
  int n  = lane & 15;
  int kb = (lane >> 4) * 8;
  union { v16h f; v8h h[2]; } u;
  u.h[0] = *(const v8h*)(srcN + n * stride + k0 + kb);
  u.h[1] = *(const v8h*)(srcN + n * stride + k0 + 16 + kb);
  return u.f;
}

// CDNA5 async bulk copy: 16B global -> LDS per lane, tracked by ASYNCcnt.
// VFLAT async encoding: VDST = LDS byte-offset VGPR, VADDR = 64-bit global addr.
static __device__ inline void async_copy_b128(void* lds_generic, const void* gptr) {
  unsigned lds_off = (unsigned)(size_t)lds_generic;   // low 32 bits = LDS offset
  asm volatile("global_load_async_to_lds_b128 %0, %1, off"
               :: "v"(lds_off), "v"(gptr) : "memory");
}
static __device__ inline void wait_asynccnt0() {
  asm volatile("s_wait_asynccnt 0x0" ::: "memory");
}

// ---------------------------------------------------------------------------
// GEMM: Y[M,N] = X[M,K] @ W[N,K]^T + bias[N]
// MODE 0: write f32 row-major [M,N];  MODE 1: f16 head-major [B,H,S,DH].
// Block: 256 thr = 8 waves, tile 128(M) x 128(N), K-chunks of 32.
// Wave (wm,wn) = (wave&3, wave>>2) owns a 32x64 subtile (2x4 WMMA frags,
// 8 WMMAs per chunk vs 6 fragment loads).
// ---------------------------------------------------------------------------
template <int MODE>
__global__ void __launch_bounds__(256)
gemm_xwt_bias(const float* __restrict__ A, const float* __restrict__ W,
              const float* __restrict__ bias, float* __restrict__ outF32,
              _Float16* __restrict__ outHead, int M, int N, int K) {
  const int S = 2048, H = 16, DH = 64;
  __shared__ _Float16 Alds[128][40];   // [m][k], stride 40 halves (16B aligned)
  __shared__ _Float16 Blds[128][40];   // [n][k]

  const int tid  = threadIdx.x;
  const int lane = tid & 31;
  const int wave = tid >> 5;
  const int wm = wave & 3, wn = wave >> 2;
  const int m0 = blockIdx.y * 128;
  const int n0 = blockIdx.x * 128;

  v8f acc[2][4] = {};

  for (int k0 = 0; k0 < K; k0 += 32) {
    __syncthreads();
    // Stage A tile 128x32 and B tile 128x32 (f32 -> f16), 16B loads / 8B stores.
    for (int i = 0; i < 4; ++i) {
      int e = tid + i * 256;                 // float4 index 0..1023
      int r = e >> 3, c4 = (e & 7) * 4;
      v4f av = *(const v4f*)(A + (size_t)(m0 + r) * K + k0 + c4);
      v4f bv = *(const v4f*)(W + (size_t)(n0 + r) * K + k0 + c4);
      v4h ah, bh;
      for (int j = 0; j < 4; ++j) { ah[j] = (_Float16)av[j]; bh[j] = (_Float16)bv[j]; }
      *(v4h*)&Alds[r][c4] = ah;
      *(v4h*)&Blds[r][c4] = bh;
    }
    __syncthreads();

    v16h a0 = load_a_frag(&Alds[wm * 32][0],      40, 0, lane);
    v16h a1 = load_a_frag(&Alds[wm * 32 + 16][0], 40, 0, lane);
    v16h b0 = load_b_frag(&Blds[wn * 64][0],      40, 0, lane);
    v16h b1 = load_b_frag(&Blds[wn * 64 + 16][0], 40, 0, lane);
    v16h b2 = load_b_frag(&Blds[wn * 64 + 32][0], 40, 0, lane);
    v16h b3 = load_b_frag(&Blds[wn * 64 + 48][0], 40, 0, lane);
    acc[0][0] = WMMA_F16(a0, b0, acc[0][0]);
    acc[0][1] = WMMA_F16(a0, b1, acc[0][1]);
    acc[0][2] = WMMA_F16(a0, b2, acc[0][2]);
    acc[0][3] = WMMA_F16(a0, b3, acc[0][3]);
    acc[1][0] = WMMA_F16(a1, b0, acc[1][0]);
    acc[1][1] = WMMA_F16(a1, b1, acc[1][1]);
    acc[1][2] = WMMA_F16(a1, b2, acc[1][2]);
    acc[1][3] = WMMA_F16(a1, b3, acc[1][3]);
  }

  // Epilogue: bias + store (C layout: m = v + 8*(lane>>4), n = lane&15).
  // MODE is compile-time: straight-line store path, no per-element branch.
  for (int r = 0; r < 2; ++r)
    for (int c = 0; c < 4; ++c)
      for (int v = 0; v < 8; ++v) {
        int m = m0 + wm * 32 + r * 16 + v + ((lane >> 4) << 3);
        int n = n0 + wn * 64 + c * 16 + (lane & 15);
        float val = acc[r][c][v] + bias[n];
        if (MODE == 0) {
          outF32[(size_t)m * N + n] = val;
        } else {
          int b = m >> 11, s = m & (S - 1);      // m = b*S + s
          int h = n >> 6,  d = n & (DH - 1);     // n = h*DH + d
          outHead[((size_t)(b * H + h) * S + s) * DH + d] = (_Float16)val;
        }
      }
}

// ---------------------------------------------------------------------------
// Causal flash attention over f16 head-major Q/K/V [B,H,S,DH], DH=64.
// Block: 128 thr = 4 waves; block owns 64 q-rows, wave owns 16 q-rows.
// K tile staged with CDNA5 async global->LDS b128 copies (ASYNCcnt).
// ---------------------------------------------------------------------------
__global__ void __launch_bounds__(128)
flash_attn(const _Float16* __restrict__ Qh, const _Float16* __restrict__ Kh,
           const _Float16* __restrict__ Vh, float* __restrict__ attnOut) {
  const int S = 2048, H = 16, DH = 64, D = 1024;
  __shared__ _Float16 Klds[64][72];        // [key][d]  (B-frag source for QK^T)
  __shared__ _Float16 Vt[64][72];          // [d][key]  (B-frag source for P.V)
  __shared__ _Float16 Plds[4][16][72];     // per-wave P tile [m][key]

  const int tid = threadIdx.x, lane = tid & 31, wave = tid >> 5;
  const int q0 = blockIdx.x * 64;
  const int bh = blockIdx.y;               // b*H + h
  const int b = bh >> 4, h = bh & 15;
  const _Float16* Qbase = Qh + (size_t)bh * S * DH;
  const _Float16* Kbase = Kh + (size_t)bh * S * DH;
  const _Float16* Vbase = Vh + (size_t)bh * S * DH;

  // Q fragments (this wave's 16 rows), A layout straight from global.
  const int qrow_lane = q0 + wave * 16 + (lane & 15);
  const int kb = (lane >> 4) * 8;
  v16h qa[2];
  for (int kc = 0; kc < 2; ++kc) {
    union { v16h f; v8h h[2]; } u;
    u.h[0] = *(const v8h*)(Qbase + (size_t)qrow_lane * DH + kc * 32 + kb);
    u.h[1] = *(const v8h*)(Qbase + (size_t)qrow_lane * DH + kc * 32 + 16 + kb);
    qa[kc] = u.f;
  }

  v8f o[4] = {};
  float m_run[8], l_run[8];
  for (int v = 0; v < 8; ++v) { m_run[v] = -1e30f; l_run[v] = 0.0f; }

  const int jmax = blockIdx.x;             // (q0+63)/64
  for (int j = 0; j <= jmax; ++j) {
    const int j0 = j * 64;
    __syncthreads();
    // Stage K tile [key][d] with async global->LDS 16B transfers.
    for (int i = 0; i < 4; ++i) {
      int e = tid + i * 128;               // 0..511 v8h units
      int r = e >> 3, cv = e & 7;
      async_copy_b128(&Klds[r][cv * 8], Kbase + (size_t)(j0 + r) * DH + cv * 8);
    }
    // Stage V transposed: Vt[d][key].
    for (int i = 0; i < 32; ++i) {
      int e = tid + i * 128;               // 0..4095
      int r = e >> 6, d = e & 63;
      Vt[d][r] = Vbase[(size_t)(j0 + r) * DH + d];
    }
    wait_asynccnt0();
    __syncthreads();

    // S = Q K^T  (M=16 rows, N=64 keys -> 4 col tiles, K=64 -> 2 chunks)
    v8f sacc[4] = {};
    for (int kc = 0; kc < 2; ++kc)
      for (int c = 0; c < 4; ++c) {
        v16h bf = load_b_frag(&Klds[c * 16][0], 72, kc * 32, lane);
        sacc[c] = WMMA_F16(qa[kc], bf, sacc[c]);
      }

    // Scale + causal mask + row-max (rows live across 16-lane half-waves).
    float rmax[8];
    for (int v = 0; v < 8; ++v) rmax[v] = -1e30f;
    for (int c = 0; c < 4; ++c)
      for (int v = 0; v < 8; ++v) {
        int key = j0 + c * 16 + (lane & 15);
        int qr  = q0 + wave * 16 + v + ((lane >> 4) << 3);
        float s = sacc[c][v] * 0.125f;     // 1/sqrt(64)
        if (key > qr) s = -1e30f;
        sacc[c][v] = s;
        rmax[v] = fmaxf(rmax[v], s);
      }
    for (int off = 1; off < 16; off <<= 1)
      for (int v = 0; v < 8; ++v)
        rmax[v] = fmaxf(rmax[v], __shfl_xor(rmax[v], off, 32));

    float corr[8], rsum[8];
    for (int v = 0; v < 8; ++v) {
      float mn = fmaxf(m_run[v], rmax[v]);
      corr[v] = __expf(m_run[v] - mn);
      m_run[v] = mn;
      rsum[v] = 0.0f;
    }
    // P = exp(S - m); spill to LDS (C layout -> A layout staging).
    for (int c = 0; c < 4; ++c)
      for (int v = 0; v < 8; ++v) {
        float p = __expf(sacc[c][v] - m_run[v]);
        rsum[v] += p;
        int mrow = v + ((lane >> 4) << 3);
        Plds[wave][mrow][c * 16 + (lane & 15)] = (_Float16)p;
      }
    for (int off = 1; off < 16; off <<= 1)
      for (int v = 0; v < 8; ++v)
        rsum[v] += __shfl_xor(rsum[v], off, 32);
    for (int v = 0; v < 8; ++v)
      l_run[v] = l_run[v] * corr[v] + rsum[v];
    for (int c = 0; c < 4; ++c)
      for (int v = 0; v < 8; ++v)
        o[c][v] *= corr[v];
    __syncthreads();

    // O += P @ V  (N=64 d-cols -> 4 tiles, K=64 keys -> 2 chunks)
    for (int kc = 0; kc < 2; ++kc) {
      v16h pa = load_a_frag(&Plds[wave][0][0], 72, kc * 32, lane);
      for (int c = 0; c < 4; ++c) {
        v16h bv = load_b_frag(&Vt[c * 16][0], 72, kc * 32, lane);
        o[c] = WMMA_F16(pa, bv, o[c]);
      }
    }
  }

  // Normalize and store f32 into [B,S,D] (D = H*DH).
  for (int c = 0; c < 4; ++c)
    for (int v = 0; v < 8; ++v) {
      int qr = q0 + wave * 16 + v + ((lane >> 4) << 3);
      int d  = c * 16 + (lane & 15);
      attnOut[((size_t)b * S + qr) * D + h * DH + d] = o[c][v] / l_run[v];
    }
}

// ---------------------------------------------------------------------------
extern "C" void kernel_launch(void* const* d_in, const int* in_sizes, int n_in,
                              void* d_out, int out_size, void* d_ws, size_t ws_size,
                              hipStream_t stream) {
  (void)in_sizes; (void)n_in; (void)out_size; (void)ws_size;
  const int B = 2, S = 2048, D = 1024, H = 16;
  const int M = B * S, N = D, K = D;

  const float* q  = (const float*)d_in[0];
  const float* k  = (const float*)d_in[1];
  const float* v  = (const float*)d_in[2];
  // d_in[3] = causal mask (bool) — computed analytically in-kernel.
  const float* Wq = (const float*)d_in[4];  const float* bq = (const float*)d_in[5];
  const float* Wk = (const float*)d_in[6];  const float* bk = (const float*)d_in[7];
  const float* Wv = (const float*)d_in[8];  const float* bv = (const float*)d_in[9];
  const float* Wo = (const float*)d_in[10]; const float* bo = (const float*)d_in[11];
  float* out = (float*)d_out;

  // Workspace: Qh/Kh/Vh f16 head-major (8 MB each) + attn f32 (16 MB) = 40 MB.
  _Float16* Qh = (_Float16*)d_ws;
  _Float16* Kh = Qh + (size_t)M * N;
  _Float16* Vh = Kh + (size_t)M * N;
  float* attn  = (float*)(Vh + (size_t)M * N);

  dim3 gg(N / 128, M / 128);   // 8 x 32 blocks
  gemm_xwt_bias<1><<<gg, 256, 0, stream>>>(q, Wq, bq, nullptr, Qh, M, N, K);
  gemm_xwt_bias<1><<<gg, 256, 0, stream>>>(k, Wk, bk, nullptr, Kh, M, N, K);
  gemm_xwt_bias<1><<<gg, 256, 0, stream>>>(v, Wv, bv, nullptr, Vh, M, N, K);

  flash_attn<<<dim3(S / 64, B * H), 128, 0, stream>>>(Qh, Kh, Vh, attn);

  gemm_xwt_bias<0><<<gg, 256, 0, stream>>>(attn, Wo, bo, out, nullptr, M, N, K);
}
